// NeighborSelector_88837103551528
// MI455X (gfx1250) — matrix operand order repaired and checked by reading, not compiled, not run on hardware
//
#include <hip/hip_runtime.h>

// ---------------------------------------------------------------------------
// CDNA5 (gfx1250, wave32) edge-MLP neighbor selector.
//   out[a, s*8+j]   = relu([x_nbr[src(a,s)] | x_agent[a]] @ W1 + b1) @ W2 + b2
//   out[a, 128:256] = 0                       (slots 16..31 unused, DEG=16)
//   out[a, 256:264] = x_agent[a] @ Wa + ba
// One agent == one 16-row WMMA A-tile (DEG == 16, edges sorted by dst).
// Each wave processes TWO agents so every W1 B-fragment fetched from L2 is
// reused for two v_wmma ops (halves L2 traffic; kernel is L2-BW limited).
// B fragments stream through a 3-deep rotating register buffer (distance-3
// software pipeline over the flattened 68-fragment sequence) so L2 latency
// is covered by in-flight WMMAs instead of s_wait_loadcnt 0 stalls.
// All LDS usage is wave-private -> no block barriers (same-wave DS in-order).
// ---------------------------------------------------------------------------

typedef __attribute__((ext_vector_type(16))) _Float16 v16h;
typedef __attribute__((ext_vector_type(8)))  _Float16 v8h;
typedef __attribute__((ext_vector_type(4)))  _Float16 v4h;
typedef __attribute__((ext_vector_type(8)))  float    v8f;

#define DEG        16
#define NBR_IN     128
#define AGENT_IN   128
#define HID        128
#define NBR_OUT    8
#define OUT_COLS   264          // MAX_NBRS*NBR_OUT + AGENT_OUT

#define WAVES      4            // waves per block
#define APW        2            // agents per wave
#define ROW_STR    264          // staging row stride (halves): 132 dw % 64 = 4 -> conflict-free, 16B aligned
#define ROW1_STR   136          // h1 row stride (halves): 68 dw % 64 = 4 -> conflict-free, 16B aligned
#define H1_TILE    (16 * ROW1_STR)            // 2176 halves per agent
#define TILE_H     (2 * H1_TILE)              // 4352 halves >= 16*ROW_STR (4224)

// workspace layout (halves)
#define W1FRAG_OFF 0            // [s:64][lane:32][hh:16] = 32768 halves (s = nt*8+kt)
#define W2FRAG_OFF 32768        // [kt:4][lane:32][hh:16] = 2048 halves
#define WS_HALVES  34816
#define NSTEP      68           // 64 layer-1 + 4 layer-2 B fragments

// ---------------------------------------------------------------------------
// Pack W1/W2 into f16 WMMA B-fragments, per-lane order (ISA 7.12.2, 16-bit B):
//   lane l holds column n = l&15, K = kbase + (l>>4)*8 + hh (hh<8), kbase + 16 + (l>>4)*8 + hh-8 (hh>=8).
// ---------------------------------------------------------------------------
__global__ void ns_prep_kernel(const float* __restrict__ W1,
                               const float* __restrict__ W2,
                               _Float16* __restrict__ wsh)
{
    const int idx = blockIdx.x * blockDim.x + threadIdx.x;
    if (idx < 32768) {
        const int hh   = idx & 15;
        const int lane = (idx >> 4) & 31;
        const int kt   = (idx >> 9) & 7;
        const int nt   = (idx >> 12) & 7;
        const int k = kt * 32 + (hh >> 3) * 16 + (lane >> 4) * 8 + (hh & 7);
        const int n = nt * 16 + (lane & 15);
        wsh[idx] = (_Float16)W1[k * HID + n];
    } else if (idx < WS_HALVES) {
        const int j    = idx - W2FRAG_OFF;
        const int hh   = j & 15;
        const int lane = (j >> 4) & 31;
        const int kt   = j >> 9;                       // 0..3
        const int k = kt * 32 + (hh >> 3) * 16 + (lane >> 4) * 8 + (hh & 7);
        const int n = lane & 15;
        wsh[idx] = (n < NBR_OUT) ? (_Float16)W2[k * NBR_OUT + n] : (_Float16)0.0f;
    }
}

// ---------------------------------------------------------------------------
// Main kernel: two agents per wave, 4 waves per block, no block barriers.
// ---------------------------------------------------------------------------
__global__ __launch_bounds__(WAVES * 32)
void ns_edge_mlp_kernel(const float* __restrict__ x_nbr,
                        const float* __restrict__ x_agent,
                        const float* __restrict__ b1,
                        const float* __restrict__ b2,
                        const int*   __restrict__ edge_src,
                        const _Float16* __restrict__ wfrag,
                        float* __restrict__ out,
                        int n_agents)
{
    __shared__ _Float16 tile[WAVES][TILE_H];         // 4 * 8704 B = 34816 B

    const int lane = threadIdx.x & 31;
    const int wave = threadIdx.x >> 5;
    const int agent0 = (blockIdx.x * WAVES + wave) * APW;
    _Float16* t = tile[wave];

    const int m  = lane & 15;     // A-layout row / C-layout column (N)
    const int kh = lane >> 4;     // A-layout K-half / C-layout M-half

    // flattened B-fragment loader: steps 0..63 = W1 (s = nt*8+kt), 64..67 = W2
    auto loadB = [&](int s) -> v16h {
        const _Float16* p = (s < 64)
            ? (wfrag + W1FRAG_OFF + (size_t)s * 512 + lane * 16)
            : (wfrag + W2FRAG_OFF + (size_t)(s - 64) * 512 + lane * 16);
        return *(const v16h*)p;
    };

    // ---- stage each agent's h = [x_nbr[src] | x_agent[a]] and pull A frags ----
    // A frag (16x32 f16, ISA 7.12.2): halves 0..7  = K kt*32 + kh*8 + 0..7
    //                                 halves 8..15 = K kt*32 + 16 + kh*8 + 0..7
    v16h afrag[APW][8];
    #pragma unroll
    for (int i = 0; i < APW; ++i) {
        int agent = agent0 + i;
        if (agent >= n_agents) agent = n_agents - 1;  // clamp (stores are guarded)

        // lane m holds src index of edge row m (lanes 16..31 duplicate)
        const int src_lane = edge_src[(size_t)agent * DEG + m];

        // phase 1: issue all gather loads back-to-back (pipelined in LOADcnt)
        float4 xa = *(const float4*)(x_agent + (size_t)agent * AGENT_IN + lane * 4);
        float4 xn[DEG];
        #pragma unroll
        for (int r = 0; r < DEG; ++r) {
            const int src = __shfl(src_lane, r, 32);
            xn[r] = *(const float4*)(x_nbr + (size_t)src * NBR_IN + lane * 4);
        }

        // phase 2: convert + store to wave-private LDS tile
        const v4h xah = { (_Float16)xa.x, (_Float16)xa.y, (_Float16)xa.z, (_Float16)xa.w };
        #pragma unroll
        for (int r = 0; r < DEG; ++r) {
            const v4h xnh = { (_Float16)xn[r].x, (_Float16)xn[r].y,
                              (_Float16)xn[r].z, (_Float16)xn[r].w };
            *(v4h*)(t + r * ROW_STR + lane * 4)          = xnh;
            *(v4h*)(t + r * ROW_STR + NBR_IN + lane * 4) = xah;
        }

        // same-wave DS ops are in-order: frag loads see the stores above
        #pragma unroll
        for (int kt = 0; kt < 8; ++kt) {
            const v8h lo = *(const v8h*)(t + m * ROW_STR + kt * 32 + kh * 8);
            const v8h hi = *(const v8h*)(t + m * ROW_STR + kt * 32 + 16 + kh * 8);
            afrag[i][kt] = __builtin_shufflevector(lo, hi,
                               0,1,2,3,4,5,6,7,8,9,10,11,12,13,14,15);
        }
    }

    // ---- prime the B pipeline (3 fragments in flight) ----
    v16h bbuf[3];
    bbuf[0] = loadB(0);
    bbuf[1] = loadB(1);
    bbuf[2] = loadB(2);

    // ---- layer 1: (16x256) @ (256x128) + b1, ReLU; each B frag feeds 2 WMMAs ----
    #pragma unroll
    for (int nt = 0; nt < 8; ++nt) {
        v8f acc0 = {};
        v8f acc1 = {};
        #pragma unroll
        for (int kt = 0; kt < 8; ++kt) {
            const int s = nt * 8 + kt;
            const v16h b = bbuf[s % 3];
            if (s + 3 < NSTEP) bbuf[s % 3] = loadB(s + 3);   // distance-3 prefetch
            acc0 = __builtin_amdgcn_wmma_f32_16x16x32_f16(false, afrag[0][kt], false, b,
                                                          (short)0, acc0, false, false);
            acc1 = __builtin_amdgcn_wmma_f32_16x16x32_f16(false, afrag[1][kt], false, b,
                                                          (short)0, acc1, false, false);
        }
        const float bias = b1[nt * 16 + m];           // C-layout: N = lane&15
        #pragma unroll
        for (int v = 0; v < 8; ++v) {                 // C-layout: M = kh*8 + v
            float r0 = acc0[v] + bias; r0 = r0 > 0.0f ? r0 : 0.0f;
            float r1 = acc1[v] + bias; r1 = r1 > 0.0f ? r1 : 0.0f;
            t[          (kh * 8 + v) * ROW1_STR + nt * 16 + m] = (_Float16)r0;
            t[H1_TILE + (kh * 8 + v) * ROW1_STR + nt * 16 + m] = (_Float16)r1;
        }
    }

    // ---- layer 2: (16x128) @ (128x8 padded to 16) + b2; B frag reused x2 ----
    v8f acc2[APW] = {};
    #pragma unroll
    for (int kt = 0; kt < 4; ++kt) {
        const int s = 64 + kt;
        const v16h b = bbuf[s % 3];                   // prefetched during layer-1 tail
        #pragma unroll
        for (int i = 0; i < APW; ++i) {
            const _Float16* h1 = t + i * H1_TILE;
            const v8h lo = *(const v8h*)(h1 + m * ROW1_STR + kt * 32 + kh * 8);
            const v8h hi = *(const v8h*)(h1 + m * ROW1_STR + kt * 32 + 16 + kh * 8);
            const v16h a2 = __builtin_shufflevector(lo, hi,
                                0,1,2,3,4,5,6,7,8,9,10,11,12,13,14,15);
            acc2[i] = __builtin_amdgcn_wmma_f32_16x16x32_f16(false, a2, false, b,
                                                             (short)0, acc2[i], false, false);
        }
    }

    // ---- store: scores (slot = tile row) + zero padding for slots 16..31 ----
    #pragma unroll
    for (int i = 0; i < APW; ++i) {
        const int agent = agent0 + i;
        if (agent >= n_agents) continue;              // guard (uniform per wave)
        float* orow = out + (size_t)agent * OUT_COLS;
        if (m < NBR_OUT) {                            // N = m valid only for n < 8
            const float bias2 = b2[m];
            #pragma unroll
            for (int v = 0; v < 8; ++v)
                orow[(kh * 8 + v) * NBR_OUT + m] = acc2[i][v] + bias2;
        }
        *(float4*)(orow + 128 + lane * 4) = make_float4(0.0f, 0.0f, 0.0f, 0.0f);
    }
}

// ---------------------------------------------------------------------------
// Own score: out[a, 256+j] = x_agent[a] @ Wa[:, j] + ba[j]   (102 MFLOP total)
// ---------------------------------------------------------------------------
__global__ void ns_own_kernel(const float* __restrict__ x_agent,
                              const float* __restrict__ Wa,
                              const float* __restrict__ ba,
                              float* __restrict__ out,
                              int n_agents)
{
    const int idx = blockIdx.x * blockDim.x + threadIdx.x;
    if (idx >= n_agents * NBR_OUT) return;
    const int j = idx & 7;
    const size_t a = (size_t)(idx >> 3);
    float acc = ba[j];
    const float* xa = x_agent + a * AGENT_IN;
    #pragma unroll 8
    for (int k = 0; k < AGENT_IN; ++k)
        acc += xa[k] * Wa[k * NBR_OUT + j];
    out[a * OUT_COLS + 256 + j] = acc;
}

// ---------------------------------------------------------------------------
extern "C" void kernel_launch(void* const* d_in, const int* in_sizes, int n_in,
                              void* d_out, int out_size, void* d_ws, size_t ws_size,
                              hipStream_t stream)
{
    const float* x_nbr    = (const float*)d_in[0];
    const float* x_agent  = (const float*)d_in[1];
    const float* W1       = (const float*)d_in[2];
    const float* b1       = (const float*)d_in[3];
    const float* W2       = (const float*)d_in[4];
    const float* b2       = (const float*)d_in[5];
    const float* Wa       = (const float*)d_in[6];
    const float* ba       = (const float*)d_in[7];
    const int*   edge_src = (const int*)d_in[8];
    float* out            = (float*)d_out;
    _Float16* wsh         = (_Float16*)d_ws;

    const int n_agents = in_sizes[1] / AGENT_IN;     // 50000

    // 1) pack W1/W2 into f16 WMMA B-fragments (coalesced per-lane order)
    ns_prep_kernel<<<(WS_HALVES + 255) / 256, 256, 0, stream>>>(W1, W2, wsh);

    // 2) edge MLP: two agents per wave
    const int blocks = (n_agents + WAVES * APW - 1) / (WAVES * APW);
    ns_edge_mlp_kernel<<<blocks, WAVES * 32, 0, stream>>>(
        x_nbr, x_agent, b1, b2, edge_src, wsh, out, n_agents);

    // 3) own scores
    const int ntot = n_agents * NBR_OUT;
    ns_own_kernel<<<(ntot + 255) / 256, 256, 0, stream>>>(x_agent, Wa, ba, out, n_agents);
}
